// MultiHead_34531537060194
// MI455X (gfx1250) — matrix operand order repaired and checked by reading, not compiled, not run on hardware
//
#include <hip/hip_runtime.h>

typedef __bf16 bf16;
typedef __attribute__((ext_vector_type(8)))  bf16  v8bf;
typedef __attribute__((ext_vector_type(16))) bf16  v16bf;
typedef __attribute__((ext_vector_type(8)))  float v8f;
typedef __attribute__((ext_vector_type(4)))  int   v4i;

#define B_ 2
#define T_ 2048
#define E_ 1024
#define H_ 16
#define LDA_S 40   // padded LDS stride (elements); 80B keeps 16B align, kills bank conflicts

// ---- CDNA5 async global->LDS copy (ASYNCcnt path), with sync fallback ----
#if defined(__HIP_DEVICE_COMPILE__) && __has_builtin(__builtin_amdgcn_global_load_async_to_lds_b128)
#define USE_ASYNC 1
#else
#define USE_ASYNC 0
#endif

#if USE_ASYNC
typedef __attribute__((address_space(1))) v4i gv4i;   // global int4
typedef __attribute__((address_space(3))) v4i lv4i;   // LDS int4
__device__ __forceinline__ void async_copy_b128(const bf16* g, bf16* l) {
    __builtin_amdgcn_global_load_async_to_lds_b128(
        (gv4i*)(unsigned long long)g,
        (lv4i*)(unsigned)(unsigned long long)l,
        0, 0);
}
#if __has_builtin(__builtin_amdgcn_s_wait_asynccnt)
#define WAIT_ASYNC() __builtin_amdgcn_s_wait_asynccnt(0)
#else
#define WAIT_ASYNC() asm volatile("s_wait_asynccnt 0x0" ::: "memory")
#endif
#else
#define WAIT_ASYNC() do {} while (0)
#endif

// ---------------------------------------------------------------- converts
__global__ void cvt_f32_bf16(const float* __restrict__ in, bf16* __restrict__ out, int n) {
    int i = blockIdx.x * blockDim.x + threadIdx.x;
    if (i < n) out[i] = (bf16)in[i];
}

// W [K=E][N=E] f32 -> Wt [N][K] bf16 (transposed so GEMM B-tiles load contiguously)
__global__ void wcvt_t_kernel(const float* __restrict__ W, bf16* __restrict__ Wt) {
    int idx = blockIdx.x * blockDim.x + threadIdx.x;
    if (idx >= E_ * E_) return;
    int k = idx >> 10;          // E_ == 1024
    int n = idx & (E_ - 1);
    Wt[(size_t)n * E_ + k] = (bf16)W[idx];
}

// V [B*T, E] bf16 -> Vt [B, H, 64, T] bf16 (per-head column-major for P@V fragments)
__global__ void vtrans_kernel(const bf16* __restrict__ V, bf16* __restrict__ Vt) {
    int idx = blockIdx.x * blockDim.x + threadIdx.x;
    if (idx >= B_ * T_ * E_) return;
    int d = idx & 63;
    int h = (idx >> 6) & (H_ - 1);
    int t = (idx >> 10) & (T_ - 1);
    int b = idx >> 21;                       // T_*E_ == 2^21
    Vt[((size_t)((b * H_ + h) * 64 + d)) * T_ + t] = V[idx];
}

// ---------------------------------------------------------------- GEMM
// C[M,N] = A[M,K] @ Bt[N,K]^T (+bias). bf16 inputs, f32 accumulate.
// Block: 256 threads (8 waves). Tile 128x128, K-step 32, double-buffered LDS
// with async global->LDS staging. Waves 4(M) x 2(N); each wave 2x4 subtiles.
__global__ __launch_bounds__(256) void gemm_bf16_kernel(
    const bf16* __restrict__ A, const bf16* __restrict__ Bt,
    bf16* __restrict__ Cb, float* __restrict__ Cf,
    const float* __restrict__ bias, int M, int N, int K)
{
    __shared__ bf16 lA[2][128 * LDA_S];   // [m][k]
    __shared__ bf16 lB[2][128 * LDA_S];   // [n][k]

    const int tid   = threadIdx.x;
    const int wave  = tid >> 5;
    const int lane  = tid & 31;
    const int lhalf = lane >> 4;
    const int l16   = lane & 15;
    const int bm    = blockIdx.x * 128;
    const int bn    = blockIdx.y * 128;
    const int wm    = (wave >> 1) * 32;
    const int wn    = (wave & 1) * 64;

    // tile-staging mapping: thread -> (row 0..127, 16-elem chunk 0/1)
    const int lr = tid >> 1;
    const int lc = (tid & 1) * 16;
    const bf16* aload = A  + (size_t)(bm + lr) * K + lc;
    const bf16* bload = Bt + (size_t)(bn + lr) * K + lc;

    auto load_tile = [&](int k0, int buf) {
        const bf16* as = aload + k0;
        const bf16* bs = bload + k0;
        bf16* ad = &lA[buf][lr * LDA_S + lc];
        bf16* bd = &lB[buf][lr * LDA_S + lc];
#if USE_ASYNC
        async_copy_b128(as,     ad);
        async_copy_b128(as + 8, ad + 8);
        async_copy_b128(bs,     bd);
        async_copy_b128(bs + 8, bd + 8);
#else
        *(v8bf*)ad       = *(const v8bf*)as;
        *(v8bf*)(ad + 8) = *(const v8bf*)(as + 8);
        *(v8bf*)bd       = *(const v8bf*)bs;
        *(v8bf*)(bd + 8) = *(const v8bf*)(bs + 8);
#endif
    };

    v8f acc[2][4];
    #pragma unroll
    for (int i = 0; i < 2; ++i)
        #pragma unroll
        for (int j = 0; j < 4; ++j)
            acc[i][j] = (v8f){0.f,0.f,0.f,0.f,0.f,0.f,0.f,0.f};

    const int nk = K >> 5;
    load_tile(0, 0);

    for (int kb = 0; kb < nk; ++kb) {
        const int cur = kb & 1;
        WAIT_ASYNC();            // tile kb landed (this wave's issues)
        __syncthreads();         // all waves' tile kb visible; buf cur^1 free
        if (kb + 1 < nk) load_tile((kb + 1) << 5, cur ^ 1);

        // B frags (32x16): lane col N=l16, K{0..15} lo-half / K{16..31} hi-half
        v16bf bfr[4];
        #pragma unroll
        for (int j = 0; j < 4; ++j) {
            const bf16* bp = &lB[cur][(wn + j * 16 + l16) * LDA_S + lhalf * 16];
            *(v8bf*)&bfr[j]       = *(const v8bf*)bp;
            *((v8bf*)&bfr[j] + 1) = *(const v8bf*)(bp + 8);
        }
        #pragma unroll
        for (int i = 0; i < 2; ++i) {
            // A frag (16x32): lane<16 row=l16 K{0..15}; lane>=16 K{16..31}
            v16bf af;
            const bf16* ap = &lA[cur][(wm + i * 16 + l16) * LDA_S + lhalf * 16];
            *(v8bf*)&af       = *(const v8bf*)ap;
            *((v8bf*)&af + 1) = *(const v8bf*)(ap + 8);
            #pragma unroll
            for (int j = 0; j < 4; ++j)
                acc[i][j] = __builtin_amdgcn_wmma_f32_16x16x32_bf16(
                    false, af, false, bfr[j], (short)0, acc[i][j], false, false);
        }
    }

    #pragma unroll
    for (int i = 0; i < 2; ++i)
        #pragma unroll
        for (int j = 0; j < 4; ++j) {
            int n  = bn + wn + j * 16 + l16;
            int mb = bm + wm + i * 16 + lhalf * 8;
            float bia = (bias != nullptr) ? bias[n] : 0.0f;
            #pragma unroll
            for (int r = 0; r < 8; ++r) {
                size_t idx = (size_t)(mb + r) * N + n;
                float v = acc[i][j][r] + bia;
                if (Cf) Cf[idx] = v;
                else    Cb[idx] = (bf16)v;
            }
        }
}

// ---------------------------------------------------------------- attention
// One wave per 16-row q-tile per (b,h). Flash-style streaming over k-tiles of 32.
__global__ __launch_bounds__(32) void attn_kernel(
    const bf16* __restrict__ Q, const bf16* __restrict__ K,
    const bf16* __restrict__ Vt, bf16* __restrict__ O, float scale)
{
    __shared__ bf16 lp[16 * LDA_S];   // P tile 16x32 staged C-layout -> A-layout

    const int lane  = threadIdx.x & 31;
    const int lhalf = lane >> 4;
    const int l16   = lane & 15;
    const int b  = blockIdx.z;
    const int h  = blockIdx.y;
    const int q0 = blockIdx.x * 16;

    // Q fragments: two 16x32 (d 0..31, 32..63), A-layout, loaded straight from global
    v16bf qf[2];
    {
        const bf16* qrow = Q + ((size_t)(b * T_ + q0 + l16) * E_ + h * 64);
        #pragma unroll
        for (int dc = 0; dc < 2; ++dc) {
            const bf16* ap = qrow + dc * 32 + lhalf * 16;
            *(v8bf*)&qf[dc]       = *(const v8bf*)ap;
            *((v8bf*)&qf[dc] + 1) = *(const v8bf*)(ap + 8);
        }
    }

    v8f o[4];
    #pragma unroll
    for (int i = 0; i < 4; ++i) o[i] = (v8f){0.f,0.f,0.f,0.f,0.f,0.f,0.f,0.f};
    float m[8], l[8];
    #pragma unroll
    for (int r = 0; r < 8; ++r) { m[r] = -1e30f; l[r] = 0.f; }

    for (int k0 = 0; k0 <= q0 + 15; k0 += 32) {
        // prefetch next k-tile (speculative)
        if (k0 + 32 <= q0 + 15) {
            __builtin_prefetch(K  + ((size_t)(b * T_ + k0 + 32 + lane) * E_ + h * 64), 0, 1);
            __builtin_prefetch(Vt + ((size_t)((b * H_ + h) * 64 + lane * 2) * T_ + k0 + 32), 0, 1);
        }

        // ---- S = Q K^T : two 16x16 score tiles, K-dim split over d (2 WMMAs each)
        v8f s[2];
        s[0] = (v8f){0.f,0.f,0.f,0.f,0.f,0.f,0.f,0.f};
        s[1] = (v8f){0.f,0.f,0.f,0.f,0.f,0.f,0.f,0.f};
        #pragma unroll
        for (int t = 0; t < 2; ++t) {
            const bf16* krow = K + ((size_t)(b * T_ + k0 + t * 16 + l16) * E_ + h * 64);
            #pragma unroll
            for (int dc = 0; dc < 2; ++dc) {
                v16bf kf;
                const bf16* bp2 = krow + dc * 32 + lhalf * 16;
                *(v8bf*)&kf       = *(const v8bf*)bp2;
                *((v8bf*)&kf + 1) = *(const v8bf*)(bp2 + 8);
                s[t] = __builtin_amdgcn_wmma_f32_16x16x32_bf16(
                    false, qf[dc], false, kf, (short)0, s[t], false, false);
            }
        }

        // ---- scale, causal mask, online softmax (rows live in 16-lane groups)
        float pa[8], pb[8];
        #pragma unroll
        for (int r = 0; r < 8; ++r) {
            int qa = q0 + lhalf * 8 + r;
            float s0 = s[0][r] * scale;
            float s1 = s[1][r] * scale;
            if (k0 + l16 > qa)      s0 = -1e30f;
            if (k0 + 16 + l16 > qa) s1 = -1e30f;
            float mt = fmaxf(s0, s1);
            mt = fmaxf(mt, __shfl_xor(mt, 1, 32));
            mt = fmaxf(mt, __shfl_xor(mt, 2, 32));
            mt = fmaxf(mt, __shfl_xor(mt, 4, 32));
            mt = fmaxf(mt, __shfl_xor(mt, 8, 32));
            float mn = fmaxf(m[r], mt);
            float sc = __expf(m[r] - mn);
            float p0 = __expf(s0 - mn);
            float p1 = __expf(s1 - mn);
            float rs = p0 + p1;
            rs += __shfl_xor(rs, 1, 32);
            rs += __shfl_xor(rs, 2, 32);
            rs += __shfl_xor(rs, 4, 32);
            rs += __shfl_xor(rs, 8, 32);
            l[r] = l[r] * sc + rs;
            m[r] = mn;
            #pragma unroll
            for (int j = 0; j < 4; ++j) o[j][r] *= sc;
            pa[r] = p0; pb[r] = p1;
        }

        // ---- P: C-layout -> A-layout via LDS
        __syncthreads();
        #pragma unroll
        for (int r = 0; r < 8; ++r) {
            lp[(lhalf * 8 + r) * LDA_S + l16]      = (bf16)pa[r];
            lp[(lhalf * 8 + r) * LDA_S + 16 + l16] = (bf16)pb[r];
        }
        __syncthreads();
        v16bf pf;
        {
            const bf16* ap = &lp[l16 * LDA_S + lhalf * 16];
            *(v8bf*)&pf       = *(const v8bf*)ap;
            *((v8bf*)&pf + 1) = *(const v8bf*)(ap + 8);
        }

        // ---- O += P @ V : 4 d-subtiles; Vt is [d][t] so fragments are contiguous
        #pragma unroll
        for (int ds = 0; ds < 4; ++ds) {
            v16bf vf;
            const bf16* vp = Vt + ((size_t)((b * H_ + h) * 64 + ds * 16 + l16) * T_
                                   + k0 + lhalf * 16);
            *(v8bf*)&vf       = *(const v8bf*)vp;
            *((v8bf*)&vf + 1) = *(const v8bf*)(vp + 8);
            o[ds] = __builtin_amdgcn_wmma_f32_16x16x32_bf16(
                false, pf, false, vf, (short)0, o[ds], false, false);
        }
    }

    // ---- normalize, store bf16
    #pragma unroll
    for (int r = 0; r < 8; ++r) {
        float inv = 1.0f / l[r];
        int row = b * T_ + q0 + lhalf * 8 + r;
        #pragma unroll
        for (int ds = 0; ds < 4; ++ds)
            O[(size_t)row * E_ + h * 64 + ds * 16 + l16] = (bf16)(o[ds][r] * inv);
    }
}

// ---------------------------------------------------------------- launch
extern "C" void kernel_launch(void* const* d_in, const int* in_sizes, int n_in,
                              void* d_out, int out_size, void* d_ws, size_t ws_size,
                              hipStream_t stream) {
    const float* x  = (const float*)d_in[0];
    const float* Wq = (const float*)d_in[1];
    const float* Wk = (const float*)d_in[2];
    const float* Wv = (const float*)d_in[3];
    const float* Wp = (const float*)d_in[4];
    const float* bp = (const float*)d_in[5];
    float* out = (float*)d_out;

    char* w = (char*)d_ws;
    const size_t MB = 1ull << 20;
    bf16* xb   = (bf16*)(w + 0 * MB);    // 8 MiB
    bf16* Wqt  = (bf16*)(w + 8 * MB);    // 2 MiB each (transposed [N][K])
    bf16* Wkt  = (bf16*)(w + 10 * MB);
    bf16* Wvt  = (bf16*)(w + 12 * MB);
    bf16* Wpt  = (bf16*)(w + 14 * MB);
    bf16* Qb   = (bf16*)(w + 16 * MB);   // 8 MiB each
    bf16* Kb   = (bf16*)(w + 24 * MB);
    bf16* Vb   = (bf16*)(w + 32 * MB);
    bf16* Vtb  = (bf16*)(w + 40 * MB);
    bf16* Ob   = (bf16*)(w + 48 * MB);   // total 56 MiB

    const int M  = B_ * T_;
    const int nx = M * E_;
    const int nw = E_ * E_;

    cvt_f32_bf16<<<(nx + 255) / 256, 256, 0, stream>>>(x, xb, nx);
    wcvt_t_kernel<<<(nw + 255) / 256, 256, 0, stream>>>(Wq, Wqt);
    wcvt_t_kernel<<<(nw + 255) / 256, 256, 0, stream>>>(Wk, Wkt);
    wcvt_t_kernel<<<(nw + 255) / 256, 256, 0, stream>>>(Wv, Wvt);
    wcvt_t_kernel<<<(nw + 255) / 256, 256, 0, stream>>>(Wp, Wpt);

    dim3 gg(M / 128, E_ / 128);
    gemm_bf16_kernel<<<gg, 256, 0, stream>>>(xb, Wqt, Qb, nullptr, nullptr, M, E_, E_);
    gemm_bf16_kernel<<<gg, 256, 0, stream>>>(xb, Wkt, Kb, nullptr, nullptr, M, E_, E_);
    gemm_bf16_kernel<<<gg, 256, 0, stream>>>(xb, Wvt, Vb, nullptr, nullptr, M, E_, E_);

    vtrans_kernel<<<(nx + 255) / 256, 256, 0, stream>>>(Vb, Vtb);

    // NOTE: reference scales by full E (1024) -> 1/32
    attn_kernel<<<dim3(T_ / 16, H_, B_), 32, 0, stream>>>(Qb, Kb, Vtb, Ob, 0.03125f);

    gemm_bf16_kernel<<<gg, 256, 0, stream>>>(Ob, Wpt, nullptr, out, bp, M, E_, E_);
}
